// MultiHeadSelfAttention_395136991401
// MI455X (gfx1250) — matrix operand order repaired and checked by reading, not compiled, not run on hardware
//
#include <hip/hip_runtime.h>

// MI455X / gfx1250 multi-head self-attention, bf16 WMMA with fp32 accumulate.
// Pipeline: qkv_proj -> flash_attn (online softmax, causal) -> out_proj.
// v2: 32x64 tiles per wave in GEMMs (B-frag reuse, 0.75 loads/wmma),
//     32-query tiles in attention (K/V frag reuse, 0.5 loads/wmma),
//     software-pipelined k-loops to overlap global loads with WMMAs.

typedef __attribute__((ext_vector_type(16))) __bf16 v16bf;
typedef __attribute__((ext_vector_type(8)))  float  v8f;

#define BATCH  2
#define SEQ    2048
#define DMODEL 1024
#define NHEAD  16
#define DHEAD  64

__device__ __forceinline__ v8f wmma_bf16(v16bf a, v16bf b, v8f c) {
  return __builtin_amdgcn_wmma_f32_16x16x32_bf16(
      /*neg_a=*/false, a, /*neg_b=*/false, b,
      /*c_mod=*/(short)0, c, /*reuse_a=*/false, /*reuse_b=*/false);
}

// ---- A-fragment: 16x32 tile, this lane holds row `row`, K per element:
//      k = (e&7) + 8*half + 16*(e>>3)  -> two contiguous 8-element runs.
__device__ __forceinline__ v16bf load_a_f32(const float* p, int ld, int row,
                                            int k0, int hf) {
  const float* r0 = p + (size_t)row * ld + k0 + hf * 8;
  const float* r1 = r0 + 16;
  v16bf f;
#pragma unroll
  for (int i = 0; i < 8; ++i) f[i] = (__bf16)r0[i];
#pragma unroll
  for (int i = 0; i < 8; ++i) f[8 + i] = (__bf16)r1[i];
  return f;
}
__device__ __forceinline__ v16bf load_a_bf16(const __bf16* p, int ld, int row,
                                             int k0, int hf) {
  const __bf16* r0 = p + (size_t)row * ld + k0 + hf * 8;
  const __bf16* r1 = r0 + 16;
  v16bf f;
#pragma unroll
  for (int i = 0; i < 8; ++i) f[i] = r0[i];
#pragma unroll
  for (int i = 0; i < 8; ++i) f[8 + i] = r1[i];
  return f;
}
// ---- B-fragment: 32x16 tile, column n sourced from row `row` of Y
//      (so wmma(A-frag(X), B-frag(Y)) = X * Y^T). k = (e&7) + 8*half + 16*(e>>3).
__device__ __forceinline__ v16bf load_b_f32(const float* p, int ld, int row,
                                            int k0, int hf) {
  const float* r0 = p + (size_t)row * ld + k0 + hf * 8;
  v16bf f;
#pragma unroll
  for (int i = 0; i < 16; ++i) f[i] = (__bf16)r0[(i & 7) + 16 * (i >> 3)];
  return f;
}
__device__ __forceinline__ v16bf load_b_bf16(const __bf16* p, int ld, int row,
                                             int k0, int hf) {
  const __bf16* r0 = p + (size_t)row * ld + k0 + hf * 8;
  v16bf f;
#pragma unroll
  for (int i = 0; i < 16; ++i) f[i] = r0[(i & 7) + 16 * (i >> 3)];
  return f;
}

// ============================================================================
// Kernel 1: Q/K/V projection.  One wave computes a 32x64 output tile with a
// software-pipelined k-loop. grid = (B*S/32, D/64, 3); z: 0->Q, 1->K, 2->V^T.
// Q,K stored [b][h][s][dh] bf16; V stored transposed [b][h][dh][s] bf16.
// ============================================================================
__global__ __launch_bounds__(32) void qkv_proj(
    const float* __restrict__ x, const float* __restrict__ WQ,
    const float* __restrict__ WK, const float* __restrict__ WV,
    __bf16* __restrict__ Q, __bf16* __restrict__ K, __bf16* __restrict__ Vt) {
  const int lane = threadIdx.x & 31;
  const int hf = lane >> 4;
  const int mrow = lane & 15;
  const int m0 = blockIdx.x * 32;   // row in flattened [B*S]
  const int n0 = blockIdx.y * 64;   // output feature base
  const int which = blockIdx.z;
  const float* W = (which == 0) ? WQ : (which == 1) ? WK : WV;

  v8f acc[8] = {};                  // acc[u*4+t]: u = row half, t = col tile

  // prime chunk k0 = 0
  v16bf aLo = load_a_f32(x, DMODEL, m0 + mrow, 0, hf);
  v16bf aHi = load_a_f32(x, DMODEL, m0 + 16 + mrow, 0, hf);
  v16bf b0 = load_b_f32(W, DMODEL, n0 + 0 * 16 + mrow, 0, hf);
  v16bf b1 = load_b_f32(W, DMODEL, n0 + 1 * 16 + mrow, 0, hf);
  v16bf b2 = load_b_f32(W, DMODEL, n0 + 2 * 16 + mrow, 0, hf);
  v16bf b3 = load_b_f32(W, DMODEL, n0 + 3 * 16 + mrow, 0, hf);

  for (int k0 = 0; k0 < DMODEL - 32; k0 += 32) {
    const int kn = k0 + 32;
    // prefetch next chunk while WMMAs below run on the current one
    v16bf naLo = load_a_f32(x, DMODEL, m0 + mrow, kn, hf);
    v16bf naHi = load_a_f32(x, DMODEL, m0 + 16 + mrow, kn, hf);
    v16bf nb0 = load_b_f32(W, DMODEL, n0 + 0 * 16 + mrow, kn, hf);
    v16bf nb1 = load_b_f32(W, DMODEL, n0 + 1 * 16 + mrow, kn, hf);
    v16bf nb2 = load_b_f32(W, DMODEL, n0 + 2 * 16 + mrow, kn, hf);
    v16bf nb3 = load_b_f32(W, DMODEL, n0 + 3 * 16 + mrow, kn, hf);

    acc[0] = wmma_bf16(aLo, b0, acc[0]);
    acc[1] = wmma_bf16(aLo, b1, acc[1]);
    acc[2] = wmma_bf16(aLo, b2, acc[2]);
    acc[3] = wmma_bf16(aLo, b3, acc[3]);
    acc[4] = wmma_bf16(aHi, b0, acc[4]);
    acc[5] = wmma_bf16(aHi, b1, acc[5]);
    acc[6] = wmma_bf16(aHi, b2, acc[6]);
    acc[7] = wmma_bf16(aHi, b3, acc[7]);

    aLo = naLo; aHi = naHi; b0 = nb0; b1 = nb1; b2 = nb2; b3 = nb3;
  }
  // epilogue chunk
  acc[0] = wmma_bf16(aLo, b0, acc[0]);
  acc[1] = wmma_bf16(aLo, b1, acc[1]);
  acc[2] = wmma_bf16(aLo, b2, acc[2]);
  acc[3] = wmma_bf16(aLo, b3, acc[3]);
  acc[4] = wmma_bf16(aHi, b0, acc[4]);
  acc[5] = wmma_bf16(aHi, b1, acc[5]);
  acc[6] = wmma_bf16(aHi, b2, acc[6]);
  acc[7] = wmma_bf16(aHi, b3, acc[7]);

#pragma unroll
  for (int u = 0; u < 2; ++u) {
#pragma unroll
    for (int t = 0; t < 4; ++t) {
      const int o = n0 + t * 16 + mrow;      // C-layout: column n = lane&15
      const int h = o >> 6, dh = o & 63;
#pragma unroll
      for (int r = 0; r < 8; ++r) {
        const int grow = m0 + u * 16 + r + 8 * hf;  // row m = r + 8*(lane/16)
        const int b = grow >> 11, s = grow & (SEQ - 1);
        const float v = acc[u * 4 + t][r];
        if (which == 2)
          Vt[(((size_t)(b * NHEAD + h)) * DHEAD + dh) * SEQ + s] = (__bf16)v;
        else {
          __bf16* dst = (which == 0) ? Q : K;
          dst[(((size_t)(b * NHEAD + h)) * SEQ + s) * DHEAD + dh] = (__bf16)v;
        }
      }
    }
  }
}

// ============================================================================
// Kernel 2: causal flash attention. One wave per (b, h, 32-query tile).
// grid = (S/32, H, B). 16 WMMA per 32-key chunk from 8 fragment loads.
// ctx written as [b][s][h*64+d] bf16 so out_proj reads it as [B*S, D].
// ============================================================================
__global__ __launch_bounds__(32) void flash_attn(
    const __bf16* __restrict__ Q, const __bf16* __restrict__ K,
    const __bf16* __restrict__ Vt, __bf16* __restrict__ ctx) {
  __shared__ __bf16 lds[32 * 32];            // P staging (C-layout -> A-layout)
  const int lane = threadIdx.x & 31;
  const int hf = lane >> 4;
  const int mrow = lane & 15;
  const int q0 = blockIdx.x * 32;
  const int h = blockIdx.y;
  const int b = blockIdx.z;
  const __bf16* Qh = Q + ((size_t)(b * NHEAD + h)) * SEQ * DHEAD;
  const __bf16* Kh = K + ((size_t)(b * NHEAD + h)) * SEQ * DHEAD;
  const __bf16* Vh = Vt + ((size_t)(b * NHEAD + h)) * DHEAD * SEQ;

  // Q fragments: aq[u*2+d] for query half u (rows q0+u*16+..) and d chunk.
  v16bf aq[4];
  aq[0] = load_a_bf16(Qh, DHEAD, q0 + mrow, 0, hf);
  aq[1] = load_a_bf16(Qh, DHEAD, q0 + mrow, 32, hf);
  aq[2] = load_a_bf16(Qh, DHEAD, q0 + 16 + mrow, 0, hf);
  aq[3] = load_a_bf16(Qh, DHEAD, q0 + 16 + mrow, 32, hf);

  float mrun[2][8], lrun[2][8];
  v8f oacc[8] = {};                          // oacc[u*4+g]
#pragma unroll
  for (int u = 0; u < 2; ++u)
#pragma unroll
    for (int r = 0; r < 8; ++r) { mrun[u][r] = -1e30f; lrun[u][r] = 0.f; }
  const float scale = 0.125f;                // 1/sqrt(64)

  // q0 is a multiple of 32, so the causal bound j0 <= q0+31 reduces to
  // j0 <= q0; max j0 = 2016 -> key reads never pass SEQ.
  for (int j0 = 0; j0 <= q0; j0 += 32) {
    // K fragments: 4 loads feed 8 score WMMAs (shared across both q halves).
    const v16bf bk00 = load_b_bf16(Kh, DHEAD, j0 + mrow, 0, hf);
    const v16bf bk01 = load_b_bf16(Kh, DHEAD, j0 + mrow, 32, hf);
    const v16bf bk10 = load_b_bf16(Kh, DHEAD, j0 + 16 + mrow, 0, hf);
    const v16bf bk11 = load_b_bf16(Kh, DHEAD, j0 + 16 + mrow, 32, hf);

    __syncthreads();                         // previous P reads done before rewrite
#pragma unroll
    for (int u = 0; u < 2; ++u) {
      v8f s0{}, s1{};
      s0 = wmma_bf16(aq[u * 2], bk00, s0);
      s0 = wmma_bf16(aq[u * 2 + 1], bk01, s0);
      s1 = wmma_bf16(aq[u * 2], bk10, s1);
      s1 = wmma_bf16(aq[u * 2 + 1], bk11, s1);

      // scale + causal mask, then fp32 online softmax (row = r + 8*hf).
      float p0[8], p1[8];
#pragma unroll
      for (int r = 0; r < 8; ++r) {
        const int qrow = q0 + u * 16 + r + 8 * hf;
        p0[r] = (j0 + mrow > qrow) ? -1e30f : s0[r] * scale;
        p1[r] = (j0 + 16 + mrow > qrow) ? -1e30f : s1[r] * scale;
      }
#pragma unroll
      for (int r = 0; r < 8; ++r) {
        float rm = fmaxf(p0[r], p1[r]);
        rm = fmaxf(rm, __shfl_xor(rm, 1));
        rm = fmaxf(rm, __shfl_xor(rm, 2));
        rm = fmaxf(rm, __shfl_xor(rm, 4));
        rm = fmaxf(rm, __shfl_xor(rm, 8));
        const float mnew = fmaxf(mrun[u][r], rm);
        const float corr = __expf(mrun[u][r] - mnew);
        const float e0 = __expf(p0[r] - mnew);
        const float e1 = __expf(p1[r] - mnew);
        float rs = e0 + e1;
        rs += __shfl_xor(rs, 1);
        rs += __shfl_xor(rs, 2);
        rs += __shfl_xor(rs, 4);
        rs += __shfl_xor(rs, 8);
        lrun[u][r] = lrun[u][r] * corr + rs;
        mrun[u][r] = mnew;
        oacc[u * 4 + 0][r] *= corr;
        oacc[u * 4 + 1][r] *= corr;
        oacc[u * 4 + 2][r] *= corr;
        oacc[u * 4 + 3][r] *= corr;
        p0[r] = e0; p1[r] = e1;
      }
      // P (C-layout) -> LDS rows u*16 + m, cols = key index.
#pragma unroll
      for (int r = 0; r < 8; ++r) {
        lds[(u * 16 + r + 8 * hf) * 32 + mrow] = (__bf16)p0[r];
        lds[(u * 16 + r + 8 * hf) * 32 + 16 + mrow] = (__bf16)p1[r];
      }
    }
    __syncthreads();

    // P A-fragments for both query halves.
    const v16bf apLo = load_a_bf16(lds, 32, mrow, 0, hf);
    const v16bf apHi = load_a_bf16(lds, 32, 16 + mrow, 0, hf);
    // V fragments: 4 loads feed 8 PV WMMAs (V^T rows contiguous in s).
#pragma unroll
    for (int g = 0; g < 4; ++g) {
      const v16bf bv = load_b_bf16(Vh, SEQ, g * 16 + mrow, j0, hf);
      oacc[g] = wmma_bf16(apLo, bv, oacc[g]);
      oacc[4 + g] = wmma_bf16(apHi, bv, oacc[4 + g]);
    }
  }

  // finalize (divide by row sum) and store ctx[b][s][h*64 + d] in bf16.
#pragma unroll
  for (int u = 0; u < 2; ++u) {
#pragma unroll
    for (int g = 0; g < 4; ++g) {
      const int dcol = h * DHEAD + g * 16 + mrow;
#pragma unroll
      for (int r = 0; r < 8; ++r) {
        const int s = q0 + u * 16 + r + 8 * hf;
        const float inv = 1.0f / lrun[u][r];
        ctx[((size_t)(b * SEQ + s)) * DMODEL + dcol] =
            (__bf16)(oacc[u * 4 + g][r] * inv);
      }
    }
  }
}

// ============================================================================
// Kernel 3: out = ctx @ WO^T (fp32 result). One wave per 32x64 tile,
// software-pipelined. grid = (B*S/32, D/64).
// ============================================================================
__global__ __launch_bounds__(32) void out_proj(
    const __bf16* __restrict__ ctx, const float* __restrict__ WO,
    float* __restrict__ out) {
  const int lane = threadIdx.x & 31;
  const int hf = lane >> 4;
  const int mrow = lane & 15;
  const int m0 = blockIdx.x * 32;
  const int n0 = blockIdx.y * 64;

  v8f acc[8] = {};
  v16bf aLo = load_a_bf16(ctx, DMODEL, m0 + mrow, 0, hf);
  v16bf aHi = load_a_bf16(ctx, DMODEL, m0 + 16 + mrow, 0, hf);
  v16bf b0 = load_b_f32(WO, DMODEL, n0 + 0 * 16 + mrow, 0, hf);
  v16bf b1 = load_b_f32(WO, DMODEL, n0 + 1 * 16 + mrow, 0, hf);
  v16bf b2 = load_b_f32(WO, DMODEL, n0 + 2 * 16 + mrow, 0, hf);
  v16bf b3 = load_b_f32(WO, DMODEL, n0 + 3 * 16 + mrow, 0, hf);

  for (int k0 = 0; k0 < DMODEL - 32; k0 += 32) {
    const int kn = k0 + 32;
    v16bf naLo = load_a_bf16(ctx, DMODEL, m0 + mrow, kn, hf);
    v16bf naHi = load_a_bf16(ctx, DMODEL, m0 + 16 + mrow, kn, hf);
    v16bf nb0 = load_b_f32(WO, DMODEL, n0 + 0 * 16 + mrow, kn, hf);
    v16bf nb1 = load_b_f32(WO, DMODEL, n0 + 1 * 16 + mrow, kn, hf);
    v16bf nb2 = load_b_f32(WO, DMODEL, n0 + 2 * 16 + mrow, kn, hf);
    v16bf nb3 = load_b_f32(WO, DMODEL, n0 + 3 * 16 + mrow, kn, hf);

    acc[0] = wmma_bf16(aLo, b0, acc[0]);
    acc[1] = wmma_bf16(aLo, b1, acc[1]);
    acc[2] = wmma_bf16(aLo, b2, acc[2]);
    acc[3] = wmma_bf16(aLo, b3, acc[3]);
    acc[4] = wmma_bf16(aHi, b0, acc[4]);
    acc[5] = wmma_bf16(aHi, b1, acc[5]);
    acc[6] = wmma_bf16(aHi, b2, acc[6]);
    acc[7] = wmma_bf16(aHi, b3, acc[7]);

    aLo = naLo; aHi = naHi; b0 = nb0; b1 = nb1; b2 = nb2; b3 = nb3;
  }
  acc[0] = wmma_bf16(aLo, b0, acc[0]);
  acc[1] = wmma_bf16(aLo, b1, acc[1]);
  acc[2] = wmma_bf16(aLo, b2, acc[2]);
  acc[3] = wmma_bf16(aLo, b3, acc[3]);
  acc[4] = wmma_bf16(aHi, b0, acc[4]);
  acc[5] = wmma_bf16(aHi, b1, acc[5]);
  acc[6] = wmma_bf16(aHi, b2, acc[6]);
  acc[7] = wmma_bf16(aHi, b3, acc[7]);

#pragma unroll
  for (int u = 0; u < 2; ++u) {
#pragma unroll
    for (int t = 0; t < 4; ++t) {
#pragma unroll
      for (int r = 0; r < 8; ++r) {
        out[((size_t)(m0 + u * 16 + r + 8 * hf)) * DMODEL + n0 + t * 16 + mrow] =
            acc[u * 4 + t][r];
      }
    }
  }
}

// ============================================================================
extern "C" void kernel_launch(void* const* d_in, const int* in_sizes, int n_in,
                              void* d_out, int out_size, void* d_ws,
                              size_t ws_size, hipStream_t stream) {
  const float* x  = (const float*)d_in[0];
  const float* WQ = (const float*)d_in[1];
  const float* WK = (const float*)d_in[2];
  const float* WV = (const float*)d_in[3];
  const float* WO = (const float*)d_in[4];
  float* out = (float*)d_out;

  const size_t n = (size_t)BATCH * SEQ * DMODEL;  // 4.19M elems
  __bf16* Q   = (__bf16*)d_ws;       // [b][h][s][dh]
  __bf16* K   = Q + n;               // [b][h][s][dh]
  __bf16* Vt  = K + n;               // [b][h][dh][s]
  __bf16* ctx = Vt + n;              // [b][s][h*64+dh]
  // total ws use: 4*n*2 bytes = 33.6 MB

  dim3 g1((BATCH * SEQ) / 32, DMODEL / 64, 3);
  qkv_proj<<<g1, 32, 0, stream>>>(x, WQ, WK, WV, Q, K, Vt);

  dim3 g2(SEQ / 32, NHEAD, BATCH);
  flash_attn<<<g2, 32, 0, stream>>>(Q, K, Vt, ctx);

  dim3 g3((BATCH * SEQ) / 32, DMODEL / 64);
  out_proj<<<g3, 32, 0, stream>>>(ctx, WO, out);
}
